// Multi_Head_Attention_37168646979686
// MI455X (gfx1250) — matrix operand order repaired and checked by reading, not compiled, not run on hardware
//
#include <hip/hip_runtime.h>

#define HID    1024
#define SEQ    2048
#define BATCH  2
#define NHEADS 16
#define HDIM   64
#define MROWS  (BATCH * SEQ)   // 4096

typedef __attribute__((ext_vector_type(16))) _Float16 v16h;
typedef __attribute__((ext_vector_type(8)))  _Float16 h8;
typedef __attribute__((ext_vector_type(4)))  _Float16 h4;
typedef __attribute__((ext_vector_type(8)))  float    v8f;
typedef __attribute__((ext_vector_type(4)))  float    f4;

// ---------- WMMA helpers (contiguous k = 16h + s per lane half, wave32) ----------
static __device__ __forceinline__ v16h wmma_load_a(const _Float16* row, int kb) {
  h8 lo = *(const h8*)(row + kb);
  h8 hi = *(const h8*)(row + kb + 8);
  v16h a;
#pragma unroll
  for (int i = 0; i < 8; ++i) { a[i] = lo[i]; a[8 + i] = hi[i]; }
  return a;
}

static __device__ __forceinline__ v8f wmma_f16(v16h a, v16h b, v8f c) {
  return __builtin_amdgcn_wmma_f32_16x16x32_f16(false, a, false, b, (short)0, c,
                                                false, false);
}

// ---------- DPP16 butterfly reductions over each 16-lane half ----------
// quad_perm[1,0,3,2]=0xB1 (xor1), quad_perm[2,3,0,1]=0x4E (xor2),
// row_half_mirror=0x141 (xor4-class), row_mirror=0x140 (xor8-class)
#define DPPF(x, ctrl)                                                         \
  __builtin_bit_cast(float, __builtin_amdgcn_update_dpp(                      \
                                0, __builtin_bit_cast(int, (x)), (ctrl), 0xf, \
                                0xf, true))

static __device__ __forceinline__ float red16_max(float x) {
  x = fmaxf(x, DPPF(x, 0xB1));
  x = fmaxf(x, DPPF(x, 0x4E));
  x = fmaxf(x, DPPF(x, 0x141));
  x = fmaxf(x, DPPF(x, 0x140));
  return x;
}
static __device__ __forceinline__ float red16_add(float x) {
  x += DPPF(x, 0xB1);
  x += DPPF(x, 0x4E);
  x += DPPF(x, 0x141);
  x += DPPF(x, 0x140);
  return x;
}

// ---------- f32 -> f16 convert ----------
__global__ void cvt_kernel(const float* __restrict__ x, _Float16* __restrict__ y,
                           int n) {
  int i = (blockIdx.x * blockDim.x + threadIdx.x) * 4;
  if (i + 3 < n) {
    f4 v = *(const f4*)(x + i);
    h4 o;
#pragma unroll
    for (int j = 0; j < 4; ++j) o[j] = (_Float16)v[j];
    *(h4*)(y + i) = o;
  }
}

// ---------- WMMA GEMM: Y = X[M,K] * W[N,K]^T + bias (+epilogue mode) ----------
// 64x32 tile per wave: 8 WMMA per K-step, B tiles reused 4x.
// MODE 0: f16 store   MODE 1: f16 transposed store (V)   MODE 2: f32 + residual
template <int MODE>
__global__ __launch_bounds__(256) void gemm_wmma_kernel(
    const _Float16* __restrict__ X, const _Float16* __restrict__ W,
    const float* __restrict__ bias, void* __restrict__ Yv,
    const float* __restrict__ R) {
  const int lane = threadIdx.x & 31;
  const int wid  = blockIdx.x * (blockDim.x >> 5) + (threadIdx.x >> 5);
  const int NTN  = HID / 32;                  // 32 col tiles
  const int tm   = (wid / NTN) * 64;
  const int tn   = (wid % NTN) * 32;

  const int ar  = lane & 15;
  const int akb = (lane >> 4) << 4;
  const int bn  = lane & 15;
  const int bkb = (lane >> 4) << 4;

  const _Float16* b0p = W + (size_t)(tn + bn) * HID + bkb;
  const _Float16* b1p = W + (size_t)(tn + 16 + bn) * HID + bkb;

  v8f c[4][2] = {};
  for (int k0 = 0; k0 < HID; k0 += 32) {
    v16h b0 = *(const v16h*)(b0p + k0);
    v16h b1 = *(const v16h*)(b1p + k0);
#pragma unroll
    for (int mt = 0; mt < 4; ++mt) {
      v16h a = wmma_load_a(X + (size_t)(tm + mt * 16 + ar) * HID + k0, akb);
      c[mt][0] = wmma_f16(a, b0, c[mt][0]);
      c[mt][1] = wmma_f16(a, b1, c[mt][1]);
    }
  }

  const int mloc = (lane >> 4) << 3;   // 0 or 8
  const int nl   = lane & 15;
#pragma unroll
  for (int mt = 0; mt < 4; ++mt) {
#pragma unroll
    for (int i = 0; i < 8; ++i) {
      int m  = tm + mt * 16 + mloc + i;
      int n0 = tn + nl, n1 = n0 + 16;
      float v0 = c[mt][0][i] + bias[n0];
      float v1 = c[mt][1][i] + bias[n1];
      if (MODE == 0) {
        _Float16* Y = (_Float16*)Yv;
        Y[(size_t)m * HID + n0] = (_Float16)v0;
        Y[(size_t)m * HID + n1] = (_Float16)v1;
      } else if (MODE == 1) {
        // Vt[b*HID + n][s]  (n = h*64+d already head-major)
        _Float16* Y = (_Float16*)Yv;
        int bi = m >> 11, s = m & (SEQ - 1);
        Y[((size_t)(bi * HID + n0)) * SEQ + s] = (_Float16)v0;
        Y[((size_t)(bi * HID + n1)) * SEQ + s] = (_Float16)v1;
      } else {
        float* Y = (float*)Yv;
        Y[(size_t)m * HID + n0] = v0 + R[(size_t)m * HID + n0];
        Y[(size_t)m * HID + n1] = v1 + R[(size_t)m * HID + n1];
      }
    }
  }
}

// ---------- Flash attention: one wave = 16 queries of one (b,h) ----------
__global__ __launch_bounds__(256) void attn_kernel(
    const _Float16* __restrict__ Q, const _Float16* __restrict__ K,
    const _Float16* __restrict__ Vt, const int* __restrict__ mask,
    _Float16* __restrict__ C) {
  __shared__ __align__(32) _Float16 pb[8 * 16 * 32];
  const int lane = threadIdx.x & 31;
  const int warp = threadIdx.x >> 5;
  _Float16* pbuf = pb + warp * (16 * 32);

  int wid = blockIdx.x * (blockDim.x >> 5) + warp;
  const int QT = SEQ / 16;          // 128 query tiles
  int qt = wid % QT;
  int bh = wid / QT;
  int h  = bh % NHEADS;
  int b  = bh / NHEADS;
  int qbase = qt * 16;

  const int ar   = lane & 15;
  const int akb  = (lane >> 4) << 4;
  const int nl   = lane & 15;
  const int bkb  = (lane >> 4) << 4;
  const int mloc = (lane >> 4) << 3;

  const _Float16* qrow = Q + ((size_t)(b * SEQ + qbase + ar)) * HID + h * HDIM;
  v16h qa0 = wmma_load_a(qrow, akb);        // d = 0..31
  v16h qa1 = wmma_load_a(qrow + 32, akb);   // d = 32..63

  float mr[8], lr[8];
#pragma unroll
  for (int i = 0; i < 8; ++i) { mr[i] = -1e30f; lr[i] = 0.f; }
  v8f acc0 = {}, acc1 = {}, acc2 = {}, acc3 = {};

  const _Float16* kbase = K + (size_t)b * SEQ * HID + h * HDIM;
  const _Float16* vbase = Vt + ((size_t)b * HID + h * HDIM) * SEQ;
  const int* mrow = mask + b * SEQ;

  for (int kt = 0; kt < SEQ; kt += 32) {
    // ---- scores: Q (16x64) . K^T  as two 16x16 tiles ----
    const _Float16* k0p = kbase + (size_t)(kt + nl) * HID + bkb;
    const _Float16* k1p = kbase + (size_t)(kt + 16 + nl) * HID + bkb;
    v8f s0 = {}, s1 = {};
    s0 = wmma_f16(qa0, *(const v16h*)(k0p), s0);
    s0 = wmma_f16(qa1, *(const v16h*)(k0p + 32), s0);
    s1 = wmma_f16(qa0, *(const v16h*)(k1p), s1);
    s1 = wmma_f16(qa1, *(const v16h*)(k1p + 32), s1);

    float bias0 = (1.0f - (float)mrow[kt + nl]) * -1e18f;
    float bias1 = (1.0f - (float)mrow[kt + 16 + nl]) * -1e18f;

    float p0[8], p1[8], tmax[8], tsum[8];
#pragma unroll
    for (int i = 0; i < 8; ++i) {
      p0[i] = s0[i] * 0.125f + bias0;   // 1/sqrt(64)
      p1[i] = s1[i] * 0.125f + bias1;
      tmax[i] = red16_max(fmaxf(p0[i], p1[i]));
    }
#pragma unroll
    for (int i = 0; i < 8; ++i) {
      float nm = fmaxf(mr[i], tmax[i]);
      float al = __expf(mr[i] - nm);
      p0[i] = __expf(p0[i] - nm);
      p1[i] = __expf(p1[i] - nm);
      tsum[i] = red16_add(p0[i] + p1[i]);
      mr[i] = nm;
      lr[i] = lr[i] * al + tsum[i];
      acc0[i] *= al; acc1[i] *= al; acc2[i] *= al; acc3[i] *= al;
    }

    // ---- stage P (C-layout) through LDS, reread in A-layout ----
#pragma unroll
    for (int i = 0; i < 8; ++i) {
      pbuf[(mloc + i) * 32 + nl]      = (_Float16)p0[i];
      pbuf[(mloc + i) * 32 + 16 + nl] = (_Float16)p1[i];
    }
    __builtin_amdgcn_wave_barrier();
    asm volatile("s_wait_dscnt 0" ::: "memory");
    v16h pa = wmma_load_a(pbuf + ar * 32, akb);

    // ---- ctx += P (16x32) . V (32x64), V transposed so loads are contiguous
    acc0 = wmma_f16(pa, *(const v16h*)(vbase + (size_t)(nl)      * SEQ + kt + bkb), acc0);
    acc1 = wmma_f16(pa, *(const v16h*)(vbase + (size_t)(16 + nl) * SEQ + kt + bkb), acc1);
    acc2 = wmma_f16(pa, *(const v16h*)(vbase + (size_t)(32 + nl) * SEQ + kt + bkb), acc2);
    acc3 = wmma_f16(pa, *(const v16h*)(vbase + (size_t)(48 + nl) * SEQ + kt + bkb), acc3);
  }

#pragma unroll
  for (int i = 0; i < 8; ++i) {
    float inv = 1.0f / lr[i];
    int m = qbase + mloc + i;
    size_t off = ((size_t)(b * SEQ + m)) * HID + h * HDIM + nl;
    C[off]      = (_Float16)(acc0[i] * inv);
    C[off + 16] = (_Float16)(acc1[i] * inv);
    C[off + 32] = (_Float16)(acc2[i] * inv);
    C[off + 48] = (_Float16)(acc3[i] * inv);
  }
}

// ---------- LayerNorm: one wave per row ----------
__global__ __launch_bounds__(256) void ln_kernel(const float* __restrict__ X,
                                                 const float* __restrict__ g,
                                                 const float* __restrict__ bt,
                                                 float* __restrict__ out) {
  int lane = threadIdx.x & 31;
  int row  = blockIdx.x * (blockDim.x >> 5) + (threadIdx.x >> 5);
  const float* r = X + (size_t)row * HID;
  float s = 0.f, ss = 0.f;
  float vals[32];
#pragma unroll
  for (int t = 0; t < 8; ++t) {
    f4 v = *(const f4*)(r + (lane + 32 * t) * 4);
#pragma unroll
    for (int j = 0; j < 4; ++j) {
      float x = v[j];
      vals[t * 4 + j] = x; s += x; ss += x * x;
    }
  }
  // reduce within 16-lane halves via DPP, then across halves via shuffle
  s = red16_add(s);  ss = red16_add(ss);
  s  += __shfl_xor(s, 16, 32);
  ss += __shfl_xor(ss, 16, 32);
  float mu = s * (1.0f / HID);
  float var = ss * (1.0f / HID) - mu * mu;
  float rs = rsqrtf(var + 1e-6f);
  float* o = out + (size_t)row * HID;
#pragma unroll
  for (int t = 0; t < 8; ++t) {
    int base = (lane + 32 * t) * 4;
    f4 res;
#pragma unroll
    for (int j = 0; j < 4; ++j)
      res[j] = (vals[t * 4 + j] - mu) * rs * g[base + j] + bt[base + j];
    *(f4*)(o + base) = res;
  }
}

extern "C" void kernel_launch(void* const* d_in, const int* in_sizes, int n_in,
                              void* d_out, int out_size, void* d_ws,
                              size_t ws_size, hipStream_t stream) {
  const float* hs  = (const float*)d_in[0];
  const int*   msk = (const int*)d_in[1];
  const float* Wq  = (const float*)d_in[2];
  const float* bq  = (const float*)d_in[3];
  const float* Wk  = (const float*)d_in[4];
  const float* bk  = (const float*)d_in[5];
  const float* Wv  = (const float*)d_in[6];
  const float* bv  = (const float*)d_in[7];
  const float* Wo  = (const float*)d_in[8];
  const float* bo  = (const float*)d_in[9];
  const float* lg  = (const float*)d_in[10];
  const float* lb  = (const float*)d_in[11];
  float* out = (float*)d_out;

  _Float16* Xh  = (_Float16*)d_ws;
  _Float16* Wqh = Xh  + (size_t)MROWS * HID;
  _Float16* Wkh = Wqh + (size_t)HID * HID;
  _Float16* Wvh = Wkh + (size_t)HID * HID;
  _Float16* Woh = Wvh + (size_t)HID * HID;
  _Float16* Qh  = Woh + (size_t)HID * HID;
  _Float16* Kh  = Qh  + (size_t)MROWS * HID;
  _Float16* Vth = Kh  + (size_t)MROWS * HID;
  _Float16* Ch  = Vth + (size_t)MROWS * HID;
  float*    O32 = (float*)(Ch + (size_t)MROWS * HID);

  const int nX = MROWS * HID, nW = HID * HID;
  cvt_kernel<<<nX / 1024, 256, 0, stream>>>(hs, Xh, nX);
  cvt_kernel<<<nW / 1024, 256, 0, stream>>>(Wq, Wqh, nW);
  cvt_kernel<<<nW / 1024, 256, 0, stream>>>(Wk, Wkh, nW);
  cvt_kernel<<<nW / 1024, 256, 0, stream>>>(Wv, Wvh, nW);
  cvt_kernel<<<nW / 1024, 256, 0, stream>>>(Wo, Woh, nW);

  const int GB = (MROWS / 64) * (HID / 32) / 8;  // 256 blocks, 8 waves each
  gemm_wmma_kernel<0><<<GB, 256, 0, stream>>>(Xh, Wqh, bq, Qh, nullptr);
  gemm_wmma_kernel<0><<<GB, 256, 0, stream>>>(Xh, Wkh, bk, Kh, nullptr);
  gemm_wmma_kernel<1><<<GB, 256, 0, stream>>>(Xh, Wvh, bv, Vth, nullptr);

  const int AB = (BATCH * NHEADS * (SEQ / 16)) / 8;  // 512 blocks
  attn_kernel<<<AB, 256, 0, stream>>>(Qh, Kh, Vth, msk, Ch);

  gemm_wmma_kernel<2><<<GB, 256, 0, stream>>>(Ch, Woh, bo, O32, hs);

  ln_kernel<<<MROWS / 8, 256, 0, stream>>>(O32, lg, lb, out);
}
